// MultiHeadLatentAttention_26877905338903
// MI455X (gfx1250) — matrix operand order repaired and checked by reading, not compiled, not run on hardware
//
#include <hip/hip_runtime.h>
#include <hip/hip_bf16.h>

// ---------------------------------------------------------------------------
// MLA (multi-head latent attention) for MI455X / gfx1250, wave32 + WMMA.
// GEMMs: TDM (tensor_load_to_lds) double-buffered LDS staging -> ds_load_b128
// fragment feeds -> v_wmma_f32_16x16x32_f16 (f32 accumulate).
// ---------------------------------------------------------------------------

typedef __attribute__((ext_vector_type(16))) _Float16 v16h;
typedef __attribute__((ext_vector_type(8)))  _Float16 v8h;
typedef __attribute__((ext_vector_type(8)))  float    v8f;
typedef __attribute__((ext_vector_type(4)))  unsigned v4u;
typedef __attribute__((ext_vector_type(8)))  int      v8i;
typedef __attribute__((ext_vector_type(4)))  int      v4i;

union V16 { v16h v; v8h h[2]; };

#define S_LEN    2048
#define HIDDEN   2048
#define NHEADS   16
#define HDIM     128
#define RDIM     64
#define KVC      512
#define QC       1024
#define HD       (NHEADS * HDIM)   // 2048
#define HR       (NHEADS * RDIM)   // 1024

#if defined(__gfx1250__) && __has_builtin(__builtin_amdgcn_tensor_load_to_lds) && \
    __has_builtin(__builtin_amdgcn_s_wait_tensorcnt)
#define USE_TDM 1
#else
#define USE_TDM 0
#endif

// ---- WMMA fragment loaders (per CDNA5 ISA 16-bit layouts, wave32) ----------
// Work for both global and LDS (generic) pointers; LDS -> ds_load_b128.

// A matrix 16x32 f16: lane l: row = m0 + (l&15); halves[0..7] = K koff..koff+7,
// halves[8..15] = K koff+16..koff+23, koff = k0 + (l>=16 ? 8 : 0).
__device__ __forceinline__ v16h load_a_frag(const _Float16* base, int lda,
                                            int m0, int k0) {
  const int lane = threadIdx.x & 31;
  const _Float16* p = base + (size_t)(m0 + (lane & 15)) * lda + k0 + ((lane >> 4) << 3);
  V16 r;
  r.h[0] = *(const v8h*)(p);
  r.h[1] = *(const v8h*)(p + 16);
  return r.v;
}

// B matrix 32x16 f16 for C = A * W^T with W stored [N,K] row-major:
// lane l: n = n0 + (l&15), halves[0..7] = K koff..koff+7, halves[8..15] = K koff+16..koff+23, koff = k0 + (l>=16 ? 8 : 0).
__device__ __forceinline__ v16h load_b_frag(const _Float16* base, int ldb,
                                            int n0, int k0) {
  const int lane = threadIdx.x & 31;
  const _Float16* p = base + (size_t)(n0 + (lane & 15)) * ldb + k0 + ((lane >> 4) << 3);
  V16 r;
  r.h[0] = *(const v8h*)(p);
  r.h[1] = *(const v8h*)(p + 16);
  return r.v;
}

__device__ __forceinline__ v8f wmma_f16(v16h a, v16h b, v8f c) {
  return __builtin_amdgcn_wmma_f32_16x16x32_f16(
      /*neg_a=*/false, a, /*neg_b=*/false, b,
      /*c_mod=*/(short)0, c, /*reuse_a=*/false, /*reuse_b=*/false);
}

// ---- TDM: 2D tile load (rows x 64 halves) into LDS -------------------------
// D# layout per CDNA5 ISA 8.3/8.4. data_size=1 (2 bytes). Tile: 64 x 128.
// LDS dest is contiguous [128][64] halves. Assumes the (single) __shared__
// block sits at LDS offset 0.

#if USE_TDM
__device__ __forceinline__ void tdm_load_tile64(const _Float16* gbase, int ldk,
                                                int rowStart, int k0, int rowsTotal,
                                                unsigned ldsByteOff) {
  unsigned long long ga =
      (unsigned long long)(const void*)gbase +
      ((unsigned long long)rowStart * (unsigned)ldk + (unsigned)k0) * 2ull;
  v4u g0;
  g0[0] = 1u;                                    // count=1, gather off
  g0[1] = ldsByteOff;                            // lds_addr
  g0[2] = (unsigned)ga;                          // global_addr[95:64]
  g0[3] = (unsigned)(ga >> 32) | (2u << 30);     // global_addr hi | type=2
  v8i g1;
  g1[0] = (1 << 16);                             // data_size=1 (2B), mask=0
  g1[1] = (int)(((unsigned)ldk & 0xFFFFu) << 16);        // tensor_dim0 lo16
  g1[2] = (int)(((unsigned)ldk >> 16) |
                (((unsigned)rowsTotal & 0xFFFFu) << 16)); // dim0 hi | dim1 lo
  g1[3] = (int)(((unsigned)rowsTotal >> 16) | (64u << 16)); // dim1 hi | tile_dim0=64
  g1[4] = 128;                                   // tile_dim1=128, tile_dim2=0
  g1[5] = ldk;                                   // tensor_dim0_stride lo32
  g1[6] = 0;                                     // stride hi16 | dim1_stride lo16
  g1[7] = 0;
  v4i z4 = {0, 0, 0, 0};
  v8i z8 = {0, 0, 0, 0, 0, 0, 0, 0};
  __builtin_amdgcn_tensor_load_to_lds(g0, g1, z4, z4, z8, 0);
}
#endif

// ---- f32 -> f16 conversion -------------------------------------------------

__global__ void cvt_f32_f16(const float* __restrict__ src,
                            _Float16* __restrict__ dst, int n) {
  int i = blockIdx.x * blockDim.x + threadIdx.x;
  int stride = gridDim.x * blockDim.x;
  for (; i < n; i += stride) dst[i] = (_Float16)src[i];
}

// ---- GEMM: C[M,N] = A[M,K] * W[N,K]^T + bias -------------------------------
// Block = 256 threads (8 waves). Output tile 128x128; each wave owns 16x128.
// K-tile = 64, double-buffered LDS (A 16KB + B 16KB per buffer = 64KB total).
// Grid = (N/128, M/128). M,N multiples of 128; K multiple of 64.

#define KT        64
#define TILE_H    (128 * KT)              // halves per A or B tile (8192)
#define TILE_B    (TILE_H * 2)            // bytes per tile (16384)

__global__ __launch_bounds__(256)
void gemm_wmma(const _Float16* __restrict__ A, const _Float16* __restrict__ W,
               const float* __restrict__ bias, float* __restrict__ Cf,
               _Float16* __restrict__ Ch, int M, int N, int K) {
  __shared__ _Float16 smem[2 * 2 * TILE_H];   // [buf][A|B][128][64] = 64KB
  const int tid   = threadIdx.x;
  const int lane  = tid & 31;
  const int wave  = tid >> 5;
  const int mTile = blockIdx.y * 128;
  const int nTile = blockIdx.x * 128;
  const int nT    = K / KT;

  v8f acc[8] = {};

#if USE_TDM
  // ---- TDM double-buffered pipeline ----
  // The TDM engine writes smem outside the compiler's alias model. Two
  // counter-measures keep the LDS fragment loads honest:
  //  (1) a dynamically-dead but compiler-visible store kills the
  //      "never-stored global => loads fold to undef" transform;
  //  (2) the escaped pointer below lets the per-iteration memory clobber
  //      alias smem, so loads cannot be forwarded/hoisted across tiles.
  if (nT < 0) smem[tid] = (_Float16)0;          // never executes at runtime
  _Float16* smem_escape = smem;
  if (tid < 32) {
    tdm_load_tile64(A, K, mTile, 0, M, 0u);
    tdm_load_tile64(W, K, nTile, 0, N, (unsigned)TILE_B);
  }
  for (int kt = 0; kt < nT; ++kt) {
    const int cur = kt & 1;
    if (tid < 32) {
      if (kt + 1 < nT) {
        const unsigned nb = (unsigned)((cur ^ 1) * 2 * TILE_B);
        tdm_load_tile64(A, K, mTile, (kt + 1) * KT, M, nb);
        tdm_load_tile64(W, K, nTile, (kt + 1) * KT, N, nb + (unsigned)TILE_B);
        __builtin_amdgcn_s_wait_tensorcnt(2);   // current tile's 2 DMAs done
      } else {
        __builtin_amdgcn_s_wait_tensorcnt(0);
      }
    }
    __syncthreads();                            // publish LDS tile to all waves
    asm volatile("" :: "v"(smem_escape) : "memory");
    const _Float16* sA = smem + cur * 2 * TILE_H;
    const _Float16* sB = sA + TILE_H;
#pragma unroll
    for (int ks = 0; ks < KT; ks += 32) {
      v16h a = load_a_frag(sA, KT, wave * 16, ks);
#pragma unroll
      for (int f = 0; f < 8; ++f) {
        v16h b = load_b_frag(sB, KT, f * 16, ks);
        acc[f] = wmma_f16(a, b, acc[f]);
      }
    }
    __syncthreads();                            // done reading before overwrite
  }
#else
  // ---- fallback: cooperative synchronous staging (single buffer) ----
  _Float16* sA = smem;
  _Float16* sB = smem + TILE_H;
  for (int kt = 0; kt < nT; ++kt) {
    const int k0 = kt * KT;
    __syncthreads();
#pragma unroll
    for (int i = 0; i < 4; ++i) {               // 1024 8-half chunks per tile
      const int c   = tid + i * 256;
      const int row = c >> 3;
      const int seg = (c & 7) << 3;
      *(v8h*)(sA + row * KT + seg) = *(const v8h*)(A + (size_t)(mTile + row) * K + k0 + seg);
      *(v8h*)(sB + row * KT + seg) = *(const v8h*)(W + (size_t)(nTile + row) * K + k0 + seg);
    }
    __syncthreads();
#pragma unroll
    for (int ks = 0; ks < KT; ks += 32) {
      v16h a = load_a_frag(sA, KT, wave * 16, ks);
#pragma unroll
      for (int f = 0; f < 8; ++f) {
        v16h b = load_b_frag(sB, KT, f * 16, ks);
        acc[f] = wmma_f16(a, b, acc[f]);
      }
    }
  }
#endif

  // ---- epilogue: D layout lane holds col n = lane&15, rows (lane>>4)*8+v ----
  const int nl = lane & 15;
  const int mb = (lane >> 4) * 8;
  const int mRow = mTile + wave * 16;
#pragma unroll
  for (int f = 0; f < 8; ++f) {
    const int col = nTile + f * 16 + nl;
    const float bv = bias ? bias[col] : 0.0f;
#pragma unroll
    for (int v = 0; v < 8; ++v) {
      const int row = mRow + mb + v;
      const float val = acc[f][v] + bv;
      if (Cf) Cf[(size_t)row * N + col] = val;
      if (Ch) Ch[(size_t)row * N + col] = (_Float16)val;
    }
  }
}

// ---- RoPE (faithful to reference: position = head index) -------------------

__global__ void rope_k(const _Float16* __restrict__ in,
                       _Float16* __restrict__ out, int npairs) {
  int i = blockIdx.x * blockDim.x + threadIdx.x;
  int stride = gridDim.x * blockDim.x;
  for (; i < npairs; i += stride) {
    const int j  = i & 31;          // freq index 0..31
    const int sh = i >> 5;          // (s*NHEADS + h)
    const int h  = sh & (NHEADS - 1);
    const int off = sh * RDIM + 2 * j;
    const float inv = __expf(-((float)(2 * j) / 64.0f) * 9.210340371976184f);
    const float ang = (float)h * inv;
    float sn, cs;
    __sincosf(ang, &sn, &cs);
    const float x1 = (float)in[off];
    const float x2 = (float)in[off + 1];
    out[off]     = (_Float16)(x1 * cs - x2 * sn);
    out[off + 1] = (_Float16)(x1 * sn + x2 * cs);
  }
}

// ---- V transpose: [S, H*128] -> [H*128, S] ---------------------------------

__global__ void transpose_v(const _Float16* __restrict__ v,
                            _Float16* __restrict__ vt, int n /* S*HD */) {
  int i = blockIdx.x * blockDim.x + threadIdx.x;
  int stride = gridDim.x * blockDim.x;
  for (; i < n; i += stride) {
    const int s = i / HD;
    const int c = i - s * HD;       // h*128 + d
    vt[(size_t)c * S_LEN + s] = v[i];
  }
}

// ---- Flash attention -------------------------------------------------------
// One wave per (head, 16 query rows). 4 waves / block.

__global__ __launch_bounds__(128)
void mla_attn(const _Float16* __restrict__ qc, const _Float16* __restrict__ qr,
              const _Float16* __restrict__ kc, const _Float16* __restrict__ kr,
              const _Float16* __restrict__ vt, _Float16* __restrict__ ctx) {
  __shared__ _Float16 lds[4 * 16 * 32];   // per-wave 16x32 P staging
  const int lane = threadIdx.x & 31;
  const int wave = threadIdx.x >> 5;
  const int tilesPerHead = S_LEN / 64;
  const int head = blockIdx.x / tilesPerHead;
  const int q0   = (blockIdx.x % tilesPerHead) * 64 + wave * 16;

  const float scale = 0.07216878364870323f;   // 1/sqrt(192)

  v16h qf[6];
#pragma unroll
  for (int i = 0; i < 4; ++i)
    qf[i] = load_a_frag(qc + head * HDIM, HD, q0, i * 32);
#pragma unroll
  for (int i = 0; i < 2; ++i)
    qf[4 + i] = load_a_frag(qr + head * RDIM, HR, q0, i * 32);

  v8f acc[8] = {};
  float m_run[8], l_run[8];
#pragma unroll
  for (int v = 0; v < 8; ++v) { m_run[v] = -3.0e38f; l_run[v] = 0.0f; }

  _Float16* myLds = lds + wave * (16 * 32);
  const int nl = lane & 15;
  const int mb = (lane >> 4) * 8;

  for (int kt = 0; kt < S_LEN; kt += 32) {
    // scores for key tiles [kt, kt+16) and [kt+16, kt+32)
    v8f s0 = {}, s1 = {};
#pragma unroll
    for (int i = 0; i < 6; ++i) {
      const _Float16* kb = (i < 4) ? (kc + head * HDIM) : (kr + head * RDIM);
      const int ld = (i < 4) ? HD : HR;
      const int kk = (i < 4) ? i * 32 : (i - 4) * 32;
      v16h b0 = load_b_frag(kb, ld, kt, kk);
      s0 = wmma_f16(qf[i], b0, s0);
      v16h b1 = load_b_frag(kb, ld, kt + 16, kk);
      s1 = wmma_f16(qf[i], b1, s1);
    }

    // online softmax; row reductions confined to 16-lane halves
    float alpha[8];
#pragma unroll
    for (int v = 0; v < 8; ++v) {
      float a0 = s0[v] * scale;
      float a1 = s1[v] * scale;
      float mv = fmaxf(a0, a1);
      mv = fmaxf(mv, __shfl_xor(mv, 1, 32));
      mv = fmaxf(mv, __shfl_xor(mv, 2, 32));
      mv = fmaxf(mv, __shfl_xor(mv, 4, 32));
      mv = fmaxf(mv, __shfl_xor(mv, 8, 32));
      const float mnew = fmaxf(m_run[v], mv);
      const float p0 = __expf(a0 - mnew);
      const float p1 = __expf(a1 - mnew);
      float rs = p0 + p1;
      rs += __shfl_xor(rs, 1, 32);
      rs += __shfl_xor(rs, 2, 32);
      rs += __shfl_xor(rs, 4, 32);
      rs += __shfl_xor(rs, 8, 32);
      const float al = __expf(m_run[v] - mnew);
      l_run[v] = l_run[v] * al + rs;
      m_run[v] = mnew;
      alpha[v] = al;
      const int row = mb + v;
      myLds[row * 32 + nl]      = (_Float16)p0;
      myLds[row * 32 + nl + 16] = (_Float16)p1;
    }

#pragma unroll
    for (int f = 0; f < 8; ++f)
#pragma unroll
      for (int v = 0; v < 8; ++v)
        acc[f][v] *= alpha[v];

    // P tile back as 16x32 A fragment (same-wave LDS, in-order)
    {
      const int arow = lane & 15;
      const int koff = (lane >> 4) << 3;
      const _Float16* lp = myLds + arow * 32 + koff;
      V16 pr;
      pr.h[0] = *(const v8h*)(lp);
      pr.h[1] = *(const v8h*)(lp + 16);
      v16h pa = pr.v;
#pragma unroll
      for (int f = 0; f < 8; ++f) {
        v16h bv = load_b_frag(vt + (size_t)head * HDIM * S_LEN, S_LEN, f * 16, kt);
        acc[f] = wmma_f16(pa, bv, acc[f]);
      }
    }
  }

#pragma unroll
  for (int f = 0; f < 8; ++f) {
#pragma unroll
    for (int v = 0; v < 8; ++v) {
      const int row = q0 + mb + v;
      const int col = head * HDIM + f * 16 + nl;
      ctx[(size_t)row * HD + col] = (_Float16)(acc[f][v] / l_run[v]);
    }
  }
}

// ---------------------------------------------------------------------------
// Host-side orchestration
// ---------------------------------------------------------------------------

extern "C" void kernel_launch(void* const* d_in, const int* in_sizes, int n_in,
                              void* d_out, int out_size, void* d_ws, size_t ws_size,
                              hipStream_t stream) {
  const float* xf    = (const float*)d_in[0];
  const float* w_kvd = (const float*)d_in[1];
  const float* b_kvd = (const float*)d_in[2];
  const float* w_kup = (const float*)d_in[3];
  const float* b_kup = (const float*)d_in[4];
  const float* w_vup = (const float*)d_in[5];
  const float* b_vup = (const float*)d_in[6];
  const float* w_kr  = (const float*)d_in[7];
  const float* b_kr  = (const float*)d_in[8];
  const float* w_qd  = (const float*)d_in[9];
  const float* b_qd  = (const float*)d_in[10];
  const float* w_qup = (const float*)d_in[11];
  const float* b_qup = (const float*)d_in[12];
  const float* w_qr  = (const float*)d_in[13];
  const float* b_qr  = (const float*)d_in[14];
  const float* w_out = (const float*)d_in[15];
  const float* b_out = (const float*)d_in[16];
  float* outp        = (float*)d_out;

  size_t ofs = 0;
  auto alloc = [&](size_t halves) -> _Float16* {
    _Float16* p = (_Float16*)((char*)d_ws + ofs);
    ofs += halves * sizeof(_Float16);
    return p;
  };

  _Float16* x16     = alloc((size_t)S_LEN * HIDDEN);
  _Float16* kvd16   = alloc((size_t)KVC * HIDDEN);
  _Float16* kup16   = alloc((size_t)HD * KVC);
  _Float16* vup16   = alloc((size_t)HD * KVC);
  _Float16* kr16    = alloc((size_t)HR * KVC);
  _Float16* qd16    = alloc((size_t)QC * HIDDEN);
  _Float16* qup16   = alloc((size_t)HD * QC);
  _Float16* qr16    = alloc((size_t)HR * QC);
  _Float16* wout16  = alloc((size_t)HIDDEN * HD);
  _Float16* kv_c16  = alloc((size_t)S_LEN * KVC);
  _Float16* q_c16   = alloc((size_t)S_LEN * QC);
  _Float16* keysc16 = alloc((size_t)S_LEN * HD);
  _Float16* vals16  = alloc((size_t)S_LEN * HD);
  _Float16* krp16   = alloc((size_t)S_LEN * HR);
  _Float16* keysr16 = alloc((size_t)S_LEN * HR);
  _Float16* qc_16   = alloc((size_t)S_LEN * HD);
  _Float16* qrp16   = alloc((size_t)S_LEN * HR);
  _Float16* qrr16   = alloc((size_t)S_LEN * HR);
  _Float16* vt16    = alloc((size_t)HD * S_LEN);
  _Float16* ctx16   = alloc((size_t)S_LEN * HD);
  (void)ws_size; (void)n_in; (void)in_sizes; (void)out_size;

  struct { const float* s; _Float16* d; int n; } cvts[] = {
    { xf,    x16,    S_LEN * HIDDEN },
    { w_kvd, kvd16,  KVC * HIDDEN   },
    { w_kup, kup16,  HD * KVC       },
    { w_vup, vup16,  HD * KVC       },
    { w_kr,  kr16,   HR * KVC       },
    { w_qd,  qd16,   QC * HIDDEN    },
    { w_qup, qup16,  HD * QC        },
    { w_qr,  qr16,   HR * QC        },
    { w_out, wout16, HIDDEN * HD    },
  };
  for (auto& c : cvts) {
    int blocks = (c.n + 255) / 256;
    if (blocks > 4096) blocks = 4096;
    cvt_f32_f16<<<blocks, 256, 0, stream>>>(c.s, c.d, c.n);
  }

  dim3 blk(256);
  auto gemm = [&](const _Float16* A, const _Float16* W, const float* bias,
                  float* Cf, _Float16* Ch, int M, int N, int K) {
    dim3 grid(N / 128, M / 128);
    gemm_wmma<<<grid, blk, 0, stream>>>(A, W, bias, Cf, Ch, M, N, K);
  };

  gemm(x16,    kvd16, b_kvd, nullptr, kv_c16,  S_LEN, KVC, HIDDEN);
  gemm(kv_c16, kup16, b_kup, nullptr, keysc16, S_LEN, HD,  KVC);
  gemm(kv_c16, vup16, b_vup, nullptr, vals16,  S_LEN, HD,  KVC);
  gemm(kv_c16, kr16,  b_kr,  nullptr, krp16,   S_LEN, HR,  KVC);
  gemm(x16,    qd16,  b_qd,  nullptr, q_c16,   S_LEN, QC,  HIDDEN);
  gemm(q_c16,  qup16, b_qup, nullptr, qc_16,   S_LEN, HD,  QC);
  gemm(q_c16,  qr16,  b_qr,  nullptr, qrp16,   S_LEN, HR,  QC);

  {
    const int npairs = S_LEN * NHEADS * (RDIM / 2);
    int blocks = (npairs + 255) / 256;
    if (blocks > 4096) blocks = 4096;
    rope_k<<<blocks, 256, 0, stream>>>(krp16, keysr16, npairs);
    rope_k<<<blocks, 256, 0, stream>>>(qrp16, qrr16, npairs);
  }

  {
    const int n = S_LEN * HD;
    int blocks = (n + 255) / 256;
    if (blocks > 4096) blocks = 4096;
    transpose_v<<<blocks, 256, 0, stream>>>(vals16, vt16, n);
  }

  {
    dim3 grid(NHEADS * (S_LEN / 64));
    mla_attn<<<grid, dim3(128), 0, stream>>>(qc_16, qrr16, keysc16, keysr16,
                                             vt16, ctx16);
  }

  gemm(ctx16, wout16, b_out, outp, nullptr, S_LEN, HIDDEN, HD);
}
